// Decoder_14362370638238
// MI455X (gfx1250) — matrix operand order repaired and checked by reading, not compiled, not run on hardware
//
#include <hip/hip_runtime.h>
#include <hip/hip_bf16.h>

// ---------------------------------------------------------------------------
// LSTM decoder: B=16, T=64, V=32000, H=1024, L=3.
// Memory-bound: weights re-read each step. Strategy: convert weights to bf16
// once (115 MB -> fits MI455X 192MB L2), run all GEMMs with
// v_wmma_f32_16x16x32_bf16 (M=16 == batch, one wave per 16-col tile).
// ---------------------------------------------------------------------------

typedef __attribute__((ext_vector_type(16))) __bf16 v16bf;
typedef __attribute__((ext_vector_type(8)))  float  v8f;

#define B_  16
#define T_  64
#define V_  32000
#define H_  1024
#define L_  3
#define G4H 4096   // 4*H

union FragBF {
  v16bf v;
  uint4 q[2];
};

__device__ __forceinline__ unsigned short f32_to_bf16u(float f) {
  union { float f; unsigned u; } v; v.f = f;
  unsigned r = v.u + 0x7FFFu + ((v.u >> 16) & 1u);   // round-to-nearest-even
  return (unsigned short)(r >> 16);
}

// ---------------------------------------------------------------------------
// One 16x16 output tile, A[16 x K] (row-major bf16, ld=ldA) times
// B^T where Brows are rows [Brow0 .. Brow0+15] of W [N x K] (row-major bf16).
// A-frag: lane<16: row m=lane, K k0..k0+15; lane>=16: row m=lane-16, K k0+16..k0+31
// B-frag: lane<16: col n=lane, K k0..k0+15; lane>=16: col n=lane-16, K k0+16..k0+31
// ---------------------------------------------------------------------------
__device__ __forceinline__ v8f gemm_tile_bf16(const unsigned short* __restrict__ A, int ldA,
                                              const unsigned short* __restrict__ W, int ldW,
                                              int Wrow0, int K, v8f acc, int lane) {
  const int r    = lane & 15;
  const int half = lane >> 4;
  const unsigned short* aPtr = A + r * ldA + half * 16;         // contiguous 16
  const unsigned short* bPtr = W + (size_t)(Wrow0 + r) * ldW + half * 16; // contiguous 16
  for (int k = 0; k < K; k += 32) {
    FragBF fa, fb;
    fa.q[0] = *(const uint4*)(aPtr + k);
    fa.q[1] = *(const uint4*)(aPtr + k + 8);
    fb.q[0] = *(const uint4*)(bPtr + k);
    fb.q[1] = *(const uint4*)(bPtr + k + 8);
    acc = __builtin_amdgcn_wmma_f32_16x16x32_bf16(false, fa.v, false, fb.v,
                                                  (short)0, acc, false, false);
  }
  return acc;
}

// ---------------------------------------------------------------------------
// Phase 0: per-(b,t) argmax over V  (reads 131MB of inp exactly once)
// ---------------------------------------------------------------------------
__global__ void dec_argmax_kernel(const float* __restrict__ inp, int* __restrict__ tok) {
  const int bt = blockIdx.x;                       // b*T + t
  const float* row = inp + (size_t)bt * V_;
  __shared__ float sv[256];
  __shared__ int   si[256];
  float best = -__builtin_inff(); int bidx = 0;
  for (int v = threadIdx.x; v < V_; v += 256) {
    float x = row[v];
    if (x > best) { best = x; bidx = v; }
  }
  sv[threadIdx.x] = best; si[threadIdx.x] = bidx;
  __syncthreads();
  for (int s = 128; s > 0; s >>= 1) {
    if (threadIdx.x < s) {
      float ov = sv[threadIdx.x + s]; int oi = si[threadIdx.x + s];
      if (ov > sv[threadIdx.x] || (ov == sv[threadIdx.x] && oi < si[threadIdx.x])) {
        sv[threadIdx.x] = ov; si[threadIdx.x] = oi;
      }
    }
    __syncthreads();
  }
  if (threadIdx.x == 0) tok[bt] = si[0];
}

// ---------------------------------------------------------------------------
// Phase 1: f32 -> bf16 weight conversion (grid-stride)
// ---------------------------------------------------------------------------
__global__ void dec_cvt_bf16_kernel(const float* __restrict__ src,
                                    unsigned short* __restrict__ dst, long long n) {
  long long i = (long long)blockIdx.x * blockDim.x + threadIdx.x;
  long long stride = (long long)gridDim.x * blockDim.x;
  for (; i < n; i += stride) dst[i] = f32_to_bf16u(src[i]);
}

// Embed tokens -> bf16 x_seq [T][B][H]
__global__ void dec_embed_kernel(const int* __restrict__ tok, const float* __restrict__ emb,
                                 unsigned short* __restrict__ xseq) {
  const int t = blockIdx.x >> 4;
  const int b = blockIdx.x & 15;
  const int tk = tok[b * T_ + t];
  const float* src = emb + (size_t)tk * H_;
  unsigned short* dst = xseq + ((size_t)t * B_ + b) * H_;
  for (int h = threadIdx.x; h < H_; h += 256) dst[h] = f32_to_bf16u(src[h]);
}

// Init f32 h/c state + bf16 h buffer A
__global__ void dec_init_state_kernel(const float* __restrict__ h0, const float* __restrict__ c0,
                                      float* __restrict__ hS, float* __restrict__ cS,
                                      unsigned short* __restrict__ hbfA) {
  int i = blockIdx.x * 256 + threadIdx.x;          // L*B*H = 49152
  if (i < L_ * B_ * H_) {
    float h = h0[i];
    hS[i] = h; cS[i] = c0[i];
    hbfA[i] = f32_to_bf16u(h);
  }
}

// ---------------------------------------------------------------------------
// Fused gates GEMM + LSTM cell for one (step, layer).
// 64 blocks x 128 threads; wave w computes gate-tile w (i/f/g/o) for the
// same 16 columns, cell nonlinearity applied in-block via LDS.
// ---------------------------------------------------------------------------
__global__ void __launch_bounds__(128)
dec_lstm_gate_kernel(const unsigned short* __restrict__ xbf,   // [16,1024] layer input
                     const unsigned short* __restrict__ hbf,   // [16,1024] prev h (read)
                     const unsigned short* __restrict__ Wih,   // [4096,1024] bf16
                     const unsigned short* __restrict__ Whh,   // [4096,1024] bf16
                     const float* __restrict__ bih,            // [4096]
                     const float* __restrict__ bhh,            // [4096]
                     float* __restrict__ hState,               // [16,1024] f32
                     float* __restrict__ cState,               // [16,1024] f32
                     unsigned short* __restrict__ hbf_out) {   // [16,1024] bf16 (write)
  __shared__ float gbuf[4][16][16];
  const int tid  = threadIdx.x;
  const int w    = tid >> 5;                       // gate index 0..3 (i,f,g,o)
  const int lane = tid & 31;
  const int jt   = blockIdx.x;                     // 0..63 column-tile
  const int N0   = w * H_ + jt * 16;               // row in W / gate col

  v8f acc = {};
  acc = gemm_tile_bf16(xbf, H_, Wih, H_, N0, H_, acc, lane);
  acc = gemm_tile_bf16(hbf, H_, Whh, H_, N0, H_, acc, lane);

  const int n     = lane & 15;
  const int mBase = (lane >> 4) * 8;
  const float bias = bih[N0 + n] + bhh[N0 + n];
#pragma unroll
  for (int r = 0; r < 8; ++r) gbuf[w][mBase + r][n] = acc[r] + bias;

  __syncthreads();

#pragma unroll
  for (int e = tid; e < 256; e += 128) {           // 16 batch x 16 cols
    const int b = e >> 4, jj = e & 15;
    const int j = jt * 16 + jj;
    float ig = gbuf[0][b][jj], fg = gbuf[1][b][jj];
    float gg = gbuf[2][b][jj], og = gbuf[3][b][jj];
    ig = 1.f / (1.f + __expf(-ig));
    fg = 1.f / (1.f + __expf(-fg));
    og = 1.f / (1.f + __expf(-og));
    gg = tanhf(gg);
    const int idx = b * H_ + j;
    float c = fg * cState[idx] + ig * gg;
    float h = og * tanhf(c);
    cState[idx]  = c;
    hState[idx]  = h;
    hbf_out[idx] = f32_to_bf16u(h);
  }
}

// ---------------------------------------------------------------------------
// Output projection: logits[b, t, :] = h_top @ W_out^T + b_out
// 250 blocks x 8 waves = 2000 tiles of 16 cols (V=32000)
// ---------------------------------------------------------------------------
__global__ void __launch_bounds__(256)
dec_proj_kernel(const unsigned short* __restrict__ xbf,    // top-layer h [16,1024]
                const unsigned short* __restrict__ Wout,   // [32000,1024] bf16
                const float* __restrict__ bout,            // [32000]
                float* __restrict__ out, int t) {
  const int tid  = threadIdx.x;
  const int lane = tid & 31;
  const int wid  = blockIdx.x * 8 + (tid >> 5);    // 0..1999
  const int N0   = wid * 16;

  v8f acc = {};
  acc = gemm_tile_bf16(xbf, H_, Wout, H_, N0, H_, acc, lane);

  const int n     = lane & 15;
  const int mBase = (lane >> 4) * 8;
  const float bias = bout[N0 + n];
#pragma unroll
  for (int r = 0; r < 8; ++r) {
    const int m = mBase + r;                       // batch index
    out[((size_t)m * T_ + t) * V_ + N0 + n] = acc[r] + bias;
  }
}

// Append final hT, cT (f32) to d_out tail
__global__ void dec_tail_kernel(const float* __restrict__ hS, const float* __restrict__ cS,
                                float* __restrict__ out) {
  const int N = L_ * B_ * H_;
  int i = blockIdx.x * 256 + threadIdx.x;
  if (i < N) {
    const size_t base = (size_t)B_ * T_ * V_;      // 32,768,000
    out[base + i]     = hS[i];
    out[base + N + i] = cS[i];
  }
}

// ---------------------------------------------------------------------------
extern "C" void kernel_launch(void* const* d_in, const int* in_sizes, int n_in,
                              void* d_out, int out_size, void* d_ws, size_t ws_size,
                              hipStream_t stream) {
  (void)in_sizes; (void)n_in; (void)out_size; (void)ws_size;
  const float* inp  = (const float*)d_in[0];
  const float* h0   = (const float*)d_in[1];
  const float* c0   = (const float*)d_in[2];
  const float* emb  = (const float*)d_in[3];
  const float* Wih  = (const float*)d_in[4];
  const float* Whh  = (const float*)d_in[5];
  const float* bih  = (const float*)d_in[6];
  const float* bhh  = (const float*)d_in[7];
  const float* Wout = (const float*)d_in[8];
  const float* bout = (const float*)d_in[9];
  float* out = (float*)d_out;

  // -------- workspace layout (256B aligned) --------
  char* ws = (char*)d_ws;
  size_t off = 0;
  auto alloc = [&](size_t bytes) { char* p = ws + off; off = (off + bytes + 255) & ~(size_t)255; return p; };

  int*            tok     = (int*)           alloc((size_t)B_ * T_ * 4);
  unsigned short* WihB    = (unsigned short*)alloc((size_t)L_ * G4H * H_ * 2);
  unsigned short* WhhB    = (unsigned short*)alloc((size_t)L_ * G4H * H_ * 2);
  unsigned short* WoutB   = (unsigned short*)alloc((size_t)V_ * H_ * 2);
  unsigned short* xseq    = (unsigned short*)alloc((size_t)T_ * B_ * H_ * 2);
  float*          hS      = (float*)         alloc((size_t)L_ * B_ * H_ * 4);
  float*          cS      = (float*)         alloc((size_t)L_ * B_ * H_ * 4);
  unsigned short* hbfA    = (unsigned short*)alloc((size_t)L_ * B_ * H_ * 2);
  unsigned short* hbfB    = (unsigned short*)alloc((size_t)L_ * B_ * H_ * 2);

  // -------- one-time prep --------
  dec_argmax_kernel<<<B_ * T_, 256, 0, stream>>>(inp, tok);
  dec_cvt_bf16_kernel<<<2048, 256, 0, stream>>>(Wih,  WihB,  (long long)L_ * G4H * H_);
  dec_cvt_bf16_kernel<<<2048, 256, 0, stream>>>(Whh,  WhhB,  (long long)L_ * G4H * H_);
  dec_cvt_bf16_kernel<<<4096, 256, 0, stream>>>(Wout, WoutB, (long long)V_ * H_);
  dec_embed_kernel<<<T_ * B_, 256, 0, stream>>>(tok, emb, xseq);
  dec_init_state_kernel<<<(L_ * B_ * H_ + 255) / 256, 256, 0, stream>>>(h0, c0, hS, cS, hbfA);

  // -------- sequential scan over T steps --------
  for (int t = 0; t < T_; ++t) {
    unsigned short* hIn  = (t & 1) ? hbfB : hbfA;  // read prev-step h
    unsigned short* hOut = (t & 1) ? hbfA : hbfB;  // write new h (no RAW race)
    const unsigned short* x = xseq + (size_t)t * B_ * H_;
    for (int l = 0; l < L_; ++l) {
      dec_lstm_gate_kernel<<<64, 128, 0, stream>>>(
          x, hIn + (size_t)l * B_ * H_,
          WihB + (size_t)l * G4H * H_, WhhB + (size_t)l * G4H * H_,
          bih + (size_t)l * G4H, bhh + (size_t)l * G4H,
          hS + (size_t)l * B_ * H_, cS + (size_t)l * B_ * H_,
          hOut + (size_t)l * B_ * H_);
      x = hOut + (size_t)l * B_ * H_;              // layer output feeds next layer
    }
    dec_proj_kernel<<<250, 256, 0, stream>>>(hOut + (size_t)(L_ - 1) * B_ * H_,
                                             WoutB, bout, out, t);
  }

  dec_tail_kernel<<<(L_ * B_ * H_ + 255) / 256, 256, 0, stream>>>(hS, cS, out);
}
